// MHA_33019708572312
// MI455X (gfx1250) — matrix operand order repaired and checked by reading, not compiled, not run on hardware
//
#include <hip/hip_runtime.h>
#include <hip/hip_bf16.h>

typedef _Float16 half_t;
typedef __attribute__((ext_vector_type(16))) _Float16 v16h;
typedef __attribute__((ext_vector_type(4)))  _Float16 v4h;
typedef __attribute__((ext_vector_type(8)))  float    v8f;

__device__ __forceinline__ v8f wmma16(v16h a, v16h b, v8f c) {
  // D = A(16x32 f16) * B(32x16 f16) + C(16x16 f32)
  return __builtin_amdgcn_wmma_f32_16x16x32_f16(false, a, false, b, (short)0, c,
                                                false, false);
}

// A-matrix fragment (16x32 f16, row-major source, leading dim `ld` in halves).
// Lane l: row = l&15; lanes 0-15 hold K 0-7 & 16-23, lanes 16-31 hold K 8-15 & 24-31.
__device__ __forceinline__ v16h load_a_frag(const half_t* p, int ld) {
  const int lane = threadIdx.x & 31;
  const half_t* r = p + (lane & 15) * ld + ((lane >> 4) << 3);
  v16h out;
  ((float4*)&out)[0] = *(const float4*)(r);       // K base .. +7
  ((float4*)&out)[1] = *(const float4*)(r + 16);  // K base+16 .. +23
  return out;
}

// B-matrix fragment (32x16 f16). Source stored as [n][k] row-major (i.e. the
// transposed operand), leading dim `ld` halves. Lane l: col n = l&15; lanes
// 0-15 hold K 0-7 & 16-23, lanes 16-31 hold K 8-15 & 24-31.
__device__ __forceinline__ v16h load_b_frag(const half_t* p, int ld) {
  const int lane = threadIdx.x & 31;
  const half_t* r = p + (lane & 15) * ld + ((lane >> 4) << 3);
  v16h out;
  ((float4*)&out)[0] = *(const float4*)(r);
  ((float4*)&out)[1] = *(const float4*)(r + 16);
  return out;
}

// Y[m,n] = sum_k A[m,k] * W[n,k] + bias[n].  M tiles on blockIdx.x (128),
// N tiles on blockIdx.y (128). K = 1024 fixed. 8 waves: 4(M) x 2(N), each wave
// computes 32x64 via 8 f32 accumulators.
template <bool A_F16, bool OUT_F16>
__global__ __launch_bounds__(256) void gemm_bias_wmma(
    const void* __restrict__ Ap, const float* __restrict__ W,
    const float* __restrict__ bias, void* __restrict__ Out) {
  constexpr int Kd = 1024;
  constexpr int LD = 40;  // 32 + 8 halves pad
  __shared__ alignas(16) half_t As[128 * LD];
  __shared__ alignas(16) half_t Bs[128 * LD];

  const int tid = threadIdx.x;
  const int lane = tid & 31;
  const int w = tid >> 5;
  const int wm = w >> 1, wn = w & 1;
  const int mBase = blockIdx.x * 128;
  const int nBase = blockIdx.y * 128;

  v8f acc[2][4] = {};

  for (int kt = 0; kt < Kd; kt += 32) {
    if (A_F16) {
      const half_t* A16 = (const half_t*)Ap;
#pragma unroll
      for (int it = 0; it < 2; ++it) {
        int s = tid + it * 256;                 // 512 slots of 8 halves
        int row = s >> 2, c = (s & 3) << 3;
        *(float4*)(As + row * LD + c) =
            *(const float4*)(A16 + (size_t)(mBase + row) * Kd + kt + c);
      }
    } else {
      const float* A32 = (const float*)Ap;
#pragma unroll
      for (int it = 0; it < 4; ++it) {
        int s = tid + it * 256;                 // 1024 slots of 4 floats
        int row = s >> 3, c = (s & 7) << 2;
        float4 v = *(const float4*)(A32 + (size_t)(mBase + row) * Kd + kt + c);
        v4h h = {(half_t)v.x, (half_t)v.y, (half_t)v.z, (half_t)v.w};
        *(v4h*)(As + row * LD + c) = h;
      }
    }
#pragma unroll
    for (int it = 0; it < 4; ++it) {
      int s = tid + it * 256;
      int row = s >> 3, c = (s & 7) << 2;
      float4 v = *(const float4*)(W + (size_t)(nBase + row) * Kd + kt + c);
      v4h h = {(half_t)v.x, (half_t)v.y, (half_t)v.z, (half_t)v.w};
      *(v4h*)(Bs + row * LD + c) = h;
    }
    __syncthreads();

    v16h a0 = load_a_frag(As + (wm * 32) * LD, LD);
    v16h a1 = load_a_frag(As + (wm * 32 + 16) * LD, LD);
#pragma unroll
    for (int ni = 0; ni < 4; ++ni) {
      v16h b = load_b_frag(Bs + (wn * 64 + ni * 16) * LD, LD);
      acc[0][ni] = wmma16(a0, b, acc[0][ni]);
      acc[1][ni] = wmma16(a1, b, acc[1][ni]);
    }
    __syncthreads();
  }

  const int rhi = (lane >> 4) << 3;
  const int cl = lane & 15;
#pragma unroll
  for (int mi = 0; mi < 2; ++mi)
#pragma unroll
    for (int ni = 0; ni < 4; ++ni)
#pragma unroll
      for (int i = 0; i < 8; ++i) {
        int row = mBase + wm * 32 + mi * 16 + i + rhi;
        int col = nBase + wn * 64 + ni * 16 + cl;
        float v = acc[mi][ni][i] + bias[col];
        if (OUT_F16)
          ((half_t*)Out)[(size_t)row * 1024 + col] = (half_t)v;
        else
          ((float*)Out)[(size_t)row * 1024 + col] = v;
      }
}

// Flash attention: one workgroup per (b, h, 128-query block); 8 waves, each
// owning 16 query rows. Online softmax; post-softmax 1/sqrt(E)=1/32 folded in.
__global__ __launch_bounds__(256) void flash_attn_wmma(
    const half_t* __restrict__ Q, const half_t* __restrict__ Kh,
    const half_t* __restrict__ Vh, half_t* __restrict__ O) {
  constexpr int Ec = 1024, dH = 64;
  __shared__ alignas(16) half_t Ks[128 * 72];       // K block, [key][d]
  __shared__ alignas(16) half_t Vs[64 * 136];       // V block transposed, [d][key]
  __shared__ alignas(16) half_t Ps[8 * 16 * 136];   // per-wave P staging, [row][key]

  const int tid = threadIdx.x;
  const int lane = tid & 31;
  const int w = tid >> 5;
  const int bh = blockIdx.y;
  const int b = bh >> 4, h = bh & 15;
  const int qb = blockIdx.x;
  const int rhi = lane >> 4;

  const size_t seqBase = (size_t)b * 1024;
  const half_t* Qbase = Q + (seqBase + qb * 128 + w * 16) * Ec + h * dH;
  v16h qf0 = load_a_frag(Qbase, Ec);        // d = 0..31
  v16h qf1 = load_a_frag(Qbase + 32, Ec);   // d = 32..63

  half_t* Pw = Ps + w * (16 * 136);

  float m_run[8], l_run[8];
  v8f oacc[4] = {};
#pragma unroll
  for (int i = 0; i < 8; ++i) { m_run[i] = -1e30f; l_run[i] = 0.f; }

  for (int jb = 0; jb < 8; ++jb) {
    {  // stage K block (direct copy)
      int row = tid >> 1;
      int off = (tid & 1) << 5;
      const half_t* src = Kh + (seqBase + jb * 128 + row) * Ec + h * dH + off;
      float4* dst = (float4*)(Ks + row * 72 + off);
#pragma unroll
      for (int j = 0; j < 4; ++j) dst[j] = ((const float4*)src)[j];
    }
    {  // stage V block transposed into [d][key]
      int row = tid >> 1;
      int off = (tid & 1) << 5;
      const half_t* src = Vh + (seqBase + jb * 128 + row) * Ec + h * dH + off;
      half_t tmp[32];
#pragma unroll
      for (int j = 0; j < 4; ++j) ((float4*)tmp)[j] = ((const float4*)src)[j];
#pragma unroll
      for (int j = 0; j < 32; ++j) Vs[(off + j) * 136 + row] = tmp[j];
    }
    __syncthreads();

    // S = Q * K^T for this wave's 16 rows x 128 keys
    v8f sacc[8];
#pragma unroll
    for (int ni = 0; ni < 8; ++ni) {
      v8f c = {};
      v16h b0 = load_b_frag(Ks + ni * 16 * 72, 72);
      v16h b1 = load_b_frag(Ks + ni * 16 * 72 + 32, 72);
      c = wmma16(qf0, b0, c);
      c = wmma16(qf1, b1, c);
      sacc[ni] = c;
    }

    // per-row block max (reduce over 8 col-tiles, then 16 lanes of half-wave)
    float bm[8];
#pragma unroll
    for (int i = 0; i < 8; ++i) {
      float m = sacc[0][i];
#pragma unroll
      for (int t = 1; t < 8; ++t) m = fmaxf(m, sacc[t][i]);
      m = fmaxf(m, __shfl_xor(m, 1, 32));
      m = fmaxf(m, __shfl_xor(m, 2, 32));
      m = fmaxf(m, __shfl_xor(m, 4, 32));
      m = fmaxf(m, __shfl_xor(m, 8, 32));
      bm[i] = m;
    }

    float alpha[8], rs[8];
#pragma unroll
    for (int i = 0; i < 8; ++i) {
      float mn = fmaxf(m_run[i], bm[i]);
      alpha[i] = __expf(m_run[i] - mn);
      m_run[i] = mn;
      rs[i] = 0.f;
    }
    // exponentiate, stage P to LDS (C-layout -> row-major)
#pragma unroll
    for (int t = 0; t < 8; ++t)
#pragma unroll
      for (int i = 0; i < 8; ++i) {
        float p = __expf(sacc[t][i] - m_run[i]);
        rs[i] += p;
        Pw[(i + 8 * rhi) * 136 + t * 16 + (lane & 15)] = (half_t)p;
      }
#pragma unroll
    for (int i = 0; i < 8; ++i) {
      float s = rs[i];
      s += __shfl_xor(s, 1, 32);
      s += __shfl_xor(s, 2, 32);
      s += __shfl_xor(s, 4, 32);
      s += __shfl_xor(s, 8, 32);
      l_run[i] = l_run[i] * alpha[i] + s;
    }
#pragma unroll
    for (int t = 0; t < 4; ++t)
#pragma unroll
      for (int i = 0; i < 8; ++i) oacc[t][i] *= alpha[i];

    // O += P * V
    v16h pf[4];
#pragma unroll
    for (int kc = 0; kc < 4; ++kc) pf[kc] = load_a_frag(Pw + kc * 32, 136);
#pragma unroll
    for (int t = 0; t < 4; ++t)
#pragma unroll
      for (int kc = 0; kc < 4; ++kc) {
        v16h bv = load_b_frag(Vs + t * 16 * 136 + kc * 32, 136);
        oacc[t] = wmma16(pf[kc], bv, oacc[t]);
      }
    __syncthreads();
  }

  // normalize: softmax sum AND post-softmax /sqrt(E)=32
#pragma unroll
  for (int t = 0; t < 4; ++t)
#pragma unroll
    for (int i = 0; i < 8; ++i) {
      int row = qb * 128 + w * 16 + i + 8 * rhi;
      int col = h * dH + t * 16 + (lane & 15);
      float v = oacc[t][i] / (l_run[i] * 32.0f);
      O[(seqBase + row) * Ec + col] = (half_t)v;
    }
}

extern "C" void kernel_launch(void* const* d_in, const int* in_sizes, int n_in,
                              void* d_out, int out_size, void* d_ws,
                              size_t ws_size, hipStream_t stream) {
  (void)in_sizes; (void)n_in; (void)out_size; (void)ws_size;
  const float* x  = (const float*)d_in[0];
  const float* Wq = (const float*)d_in[1];
  const float* bq = (const float*)d_in[2];
  const float* Wk = (const float*)d_in[3];
  const float* bk = (const float*)d_in[4];
  const float* Wv = (const float*)d_in[5];
  const float* bv = (const float*)d_in[6];
  const float* Wo = (const float*)d_in[7];
  const float* bo = (const float*)d_in[8];

  const size_t MN = (size_t)4096 * 1024;
  half_t* Qh = (half_t*)d_ws;
  half_t* Kh = Qh + MN;
  half_t* Vh = Kh + MN;
  half_t* Oh = Vh + MN;  // 32 MB of workspace total

  dim3 g(32, 8), blk(256);
  gemm_bias_wmma<false, true><<<g, blk, 0, stream>>>(x, Wq, bq, Qh);
  gemm_bias_wmma<false, true><<<g, blk, 0, stream>>>(x, Wk, bk, Kh);
  gemm_bias_wmma<false, true><<<g, blk, 0, stream>>>(x, Wv, bv, Vh);
  flash_attn_wmma<<<dim3(8, 64), blk, 0, stream>>>(Qh, Kh, Vh, Oh);
  gemm_bias_wmma<true, false><<<g, blk, 0, stream>>>(Oh, Wo, bo, d_out);
}
